// MPNN_36146444763728
// MI455X (gfx1250) — matrix operand order repaired and checked by reading, not compiled, not run on hardware
//
#include <hip/hip_runtime.h>

typedef __attribute__((ext_vector_type(16))) _Float16 v16h;
typedef __attribute__((ext_vector_type(8)))  _Float16 v8h;
typedef __attribute__((ext_vector_type(8)))  float    v8f;

#define DEV_INLINE __device__ __forceinline__

DEV_INLINE float sigf(float x) { return 1.0f / (1.0f + __expf(-x)); }

// ---------------- elementwise / prep kernels ----------------

__global__ void k_zero_f32(float* p, int n) {
  int i = blockIdx.x * 256 + threadIdx.x;
  if (i < n) p[i] = 0.0f;
}
__global__ void k_zero_f16(_Float16* p, int n) {
  int i = blockIdx.x * 256 + threadIdx.x;
  if (i < n) p[i] = (_Float16)0.0f;
}
__global__ void k_copy_f32(float* d, const float* s, int n) {
  int i = blockIdx.x * 256 + threadIdx.x;
  if (i < n) d[i] = s[i];
}
// f32 (rows x cols) -> f16 (rows x colsp), zero padded
__global__ void k_cvt_pad_f16(const float* __restrict__ src, _Float16* __restrict__ dst,
                              int rows, int cols, int colsp) {
  int i = blockIdx.x * 256 + threadIdx.x;
  int total = rows * colsp;
  if (i >= total) return;
  int r = i / colsp, c = i - r * colsp;
  dst[i] = (c < cols) ? (_Float16)src[r * cols + c] : (_Float16)0.0f;
}
__global__ void k_add_vec(const float* a, const float* b, float* c, int n) {
  int i = blockIdx.x * 256 + threadIdx.x;
  if (i < n) c[i] = a[i] + b[i];
}
// W2T[o][i*14+k] = We[(i*64+o)*13 + k] for k<13 ; = be[i*64+o] for k==13   (f16, 64 x 896)
__global__ void k_build_w2t(const float* __restrict__ We, const float* __restrict__ be,
                            _Float16* __restrict__ w2t) {
  int i = blockIdx.x * 256 + threadIdx.x;
  if (i >= 64 * 896) return;
  int o = i / 896;
  int rem = i - o * 896;
  int ii = rem / 14, k = rem - ii * 14;
  float v = (k < 13) ? We[(ii * 64 + o) * 13 + k] : be[ii * 64 + o];
  w2t[i] = (_Float16)v;
}

// ---------------- generic WMMA GEMM:  C[m][n] = sum_k A16[m][k] * W16[n][k] (+bias / +=C) ----
// One wave computes a 16(rows) x 64(cols) tile: 1 A-fragment load amortized over 4 WMMAs.
// A16: M x Kp (f16, Kp % 32 == 0), W16: N x Kp (f16), C: M x N (f32); N % 64 == 0.
// flags: 1 = ReLU on output, 2 = accumulate (init acc from C, ignore bias)
__global__ void k_wmma_gemm(const _Float16* __restrict__ A, const _Float16* __restrict__ W,
                            const float* __restrict__ bias, float* __restrict__ C,
                            int M, int N, int Kp, int flags) {
  int l = threadIdx.x;            // 0..31
  int ntile = blockIdx.x;         // 64-wide column tile
  int mtile = blockIdx.y;
  int lrow = l & 15;
  int hi = l >> 4;                // 0 or 1
  int arow = mtile * 16 + lrow;
  if (arow >= M) arow = M - 1;    // clamp (stores are guarded)
  int colbase = ntile * 64 + lrow;

  v8f acc[4];
  if (flags & 2) {
#pragma unroll
    for (int s = 0; s < 4; ++s)
#pragma unroll
      for (int v = 0; v < 8; ++v) {
        int m = mtile * 16 + hi * 8 + v;
        if (m >= M) m = M - 1;
        acc[s][v] = C[m * N + colbase + s * 16];
      }
  } else {
#pragma unroll
    for (int s = 0; s < 4; ++s) {
      float bv = bias ? bias[colbase + s * 16] : 0.0f;
#pragma unroll
      for (int v = 0; v < 8; ++v) acc[s][v] = bv;
    }
  }

  const _Float16* ar = A + (size_t)arow * Kp;
  const _Float16* br0 = W + (size_t)(colbase +  0) * Kp;
  const _Float16* br1 = W + (size_t)(colbase + 16) * Kp;
  const _Float16* br2 = W + (size_t)(colbase + 32) * Kp;
  const _Float16* br3 = W + (size_t)(colbase + 48) * Kp;

  for (int kc = 0; kc < Kp; kc += 32) {
    // A fragment: lane holds row (l&15), K = hi*16 + [0..15]
    v16h a = *(const v16h*)(ar + kc + hi * 16);
    // B fragments: lane holds col (l&15) of each 16-wide subtile, K = hi*16 + [0..15]
    v16h b0 = *(const v16h*)(br0 + kc + hi * 16);
    v16h b1 = *(const v16h*)(br1 + kc + hi * 16);
    v16h b2 = *(const v16h*)(br2 + kc + hi * 16);
    v16h b3 = *(const v16h*)(br3 + kc + hi * 16);
    acc[0] = __builtin_amdgcn_wmma_f32_16x16x32_f16(false, a, false, b0, (short)0, acc[0], false, false);
    acc[1] = __builtin_amdgcn_wmma_f32_16x16x32_f16(false, a, false, b1, (short)0, acc[1], false, false);
    acc[2] = __builtin_amdgcn_wmma_f32_16x16x32_f16(false, a, false, b2, (short)0, acc[2], false, false);
    acc[3] = __builtin_amdgcn_wmma_f32_16x16x32_f16(false, a, false, b3, (short)0, acc[3], false, false);
  }

#pragma unroll
  for (int s = 0; s < 4; ++s)
#pragma unroll
    for (int v = 0; v < 8; ++v) {
      int m = mtile * 16 + hi * 8 + v;
      if (m < M) {
        float r = acc[s][v];
        if (flags & 1) r = r > 0.0f ? r : 0.0f;
        C[m * N + colbase + s * 16] = r;
      }
    }
}

// ---------------- graph-specific kernels ----------------

__global__ void k_deg(const int* __restrict__ dst, float* __restrict__ deg, int E) {
  int e = blockIdx.x * 256 + threadIdx.x;
  if (e < E) atomicAdd(&deg[dst[e]], 1.0f);
}

__global__ void k_gate(const float* __restrict__ h0, const int* __restrict__ src,
                       const int* __restrict__ dst, const float* __restrict__ Wg,
                       const float* __restrict__ bg, const float* __restrict__ deg,
                       float* __restrict__ egate, int E) {
  int e = blockIdx.x * 256 + threadIdx.x;
  if (e >= E) return;
  int s = src[e], d = dst[e];
  const float* hd = h0 + d * 64;
  const float* hs = h0 + s * 64;
  float acc = bg[0];
#pragma unroll 8
  for (int j = 0; j < 64; ++j) acc += hd[j] * Wg[j] + hs[j] * Wg[64 + j];
  float t = tanhf(acc);
  egate[e] = 0.3f + t * deg[d] * deg[s];
}

// Fused NNConv message: per block -> 32 edges.
// Builds X[e][i*14+k'] = h[src[e],i] * efeat_ext[e,k'] in LDS (f16), then
// 8 waves each do one 16(edge) x 16(out) WMMA tile over K=896, scatter-atomic to agg[dst].
__global__ void k_edge_msg(const float* __restrict__ h, const float* __restrict__ edge_attr,
                           const float* __restrict__ egate, const int* __restrict__ src,
                           const int* __restrict__ dst, const _Float16* __restrict__ W2T,
                           float* __restrict__ agg, int E) {
  __shared__ __align__(32) _Float16 X[32 * 896];
  __shared__ float EF[32 * 14];
  int tid = threadIdx.x;          // 256 threads
  int ebase = blockIdx.x * 32;

  for (int i = tid; i < 32 * 14; i += 256) {
    int e = i / 14, k = i - e * 14;
    int ge = ebase + e;
    float v;
    if (k < 12)      v = edge_attr[ge * 12 + k];
    else if (k == 12) v = egate[ge];
    else              v = 1.0f;
    EF[i] = v;
  }
  __syncthreads();

  {
    int e = tid >> 3;             // 0..31
    int part = tid & 7;           // 0..7  (covers i = part*8 .. part*8+7)
    int ge = ebase + e;
    const float* hs = h + (size_t)src[ge] * 64;
    _Float16* xr = X + e * 896 + part * 112;
    const float* efr = EF + e * 14;
#pragma unroll
    for (int ii = 0; ii < 8; ++ii) {
      float hv = hs[part * 8 + ii];
#pragma unroll
      for (int k = 0; k < 14; ++k) xr[ii * 14 + k] = (_Float16)(hv * efr[k]);
    }
  }
  __syncthreads();

  int w = tid >> 5;               // wave 0..7
  int l = tid & 31;
  int et = w >> 2;                // edge sub-tile 0/1
  int ot = w & 3;                 // output col tile 0..3
  int lrow = l & 15;
  int hi = l >> 4;

  v8f acc;
#pragma unroll
  for (int v = 0; v < 8; ++v) acc[v] = 0.0f;

  const _Float16* xrow = X + (et * 16 + lrow) * 896;
  const _Float16* brow = W2T + (ot * 16 + lrow) * 896;
  for (int kc = 0; kc < 896; kc += 32) {
    v16h a = *(const v16h*)(xrow + kc + hi * 16);
    v16h b = *(const v16h*)(brow + kc + hi * 16);
    acc = __builtin_amdgcn_wmma_f32_16x16x32_f16(false, a, false, b, (short)0, acc, false, false);
  }

  int o = ot * 16 + lrow;
#pragma unroll
  for (int v = 0; v < 8; ++v) {
    int ge = ebase + et * 16 + hi * 8 + v;
    if (ge < E) atomicAdd(&agg[(size_t)dst[ge] * 64 + o], acc[v]);
  }
}

// x16 = f16( relu(agg + nnconv_bias) )
__global__ void k_relu_x16(const float* __restrict__ agg, const float* __restrict__ bias,
                           _Float16* __restrict__ x16, int total) {
  int i = blockIdx.x * 256 + threadIdx.x;
  if (i >= total) return;
  float v = agg[i] + bias[i & 63];
  x16[i] = (_Float16)(v > 0.0f ? v : 0.0f);
}

// GRU cell update (torch gate order r,z,n); gi/gh already include biases
__global__ void k_gru(const float* __restrict__ gi, const float* __restrict__ gh,
                      float* __restrict__ hid, _Float16* __restrict__ hid16, int total) {
  int i = blockIdx.x * 256 + threadIdx.x;
  if (i >= total) return;
  int n = i >> 6, j = i & 63;
  const float* gin = gi + n * 192;
  const float* ghn = gh + n * 192;
  float r = sigf(gin[j] + ghn[j]);
  float z = sigf(gin[64 + j] + ghn[64 + j]);
  float ng = tanhf(gin[128 + j] + r * ghn[128 + j]);
  float hv = hid[i];
  float nh = (1.0f - z) * ng + z * hv;
  hid[i] = nh;
  hid16[i] = (_Float16)nh;
}

// LSTM cell (gate order i,f,g,o), D=128
__global__ void k_lstm(const float* __restrict__ g4, float* __restrict__ cl,
                       float* __restrict__ hl, _Float16* __restrict__ hl16, int total) {
  int i = blockIdx.x * 256 + threadIdx.x;
  if (i >= total) return;
  int b = i >> 7, d = i & 127;
  const float* g = g4 + b * 512;
  float ig = sigf(g[d]);
  float fg = sigf(g[128 + d]);
  float gg = tanhf(g[256 + d]);
  float og = sigf(g[384 + d]);
  float c = fg * cl[i] + ig * gg;
  float hv = og * tanhf(c);
  cl[i] = c;
  hl[i] = hv;
  hl16[i] = (_Float16)hv;
}

// Set2Set attention + readout, one block per graph (40 contiguous nodes), 64 threads
__global__ void k_attn(const float* __restrict__ h0, const float* __restrict__ h,
                       const float* __restrict__ hl, float* __restrict__ qstar,
                       _Float16* __restrict__ qstar16) {
  int b = blockIdx.x;
  int t = threadIdx.x;            // 64
  __shared__ float sc[64];
  __shared__ float mx, den;
  const float* q = hl + b * 128;
  int n0 = b * 40;
  if (t < 40) {
    const float* f0 = h0 + (size_t)(n0 + t) * 64;
    const float* f1 = h  + (size_t)(n0 + t) * 64;
    float acc = 0.0f;
#pragma unroll 8
    for (int d = 0; d < 64; ++d) acc += f0[d] * q[d] + f1[d] * q[64 + d];
    sc[t] = acc;
  }
  __syncthreads();
  if (t == 0) {
    float m = -1e30f;
    for (int i = 0; i < 40; ++i) m = fmaxf(m, sc[i]);
    mx = m;
  }
  __syncthreads();
  if (t < 40) sc[t] = __expf(sc[t] - mx);
  __syncthreads();
  if (t == 0) {
    float s = 0.0f;
    for (int i = 0; i < 40; ++i) s += sc[i];
    den = s;
  }
  __syncthreads();
  for (int d = t; d < 128; d += 64) {
    float acc = 0.0f;
    for (int i = 0; i < 40; ++i) {
      float fv = (d < 64) ? h0[(size_t)(n0 + i) * 64 + d] : h[(size_t)(n0 + i) * 64 + (d - 64)];
      acc += (sc[i] / den) * fv;
    }
    float qv = q[d];
    qstar[b * 256 + d] = qv;
    qstar[b * 256 + 128 + d] = acc;
    qstar16[b * 256 + d] = (_Float16)qv;
    qstar16[b * 256 + 128 + d] = (_Float16)acc;
  }
}

__global__ void k_prelu(float* __restrict__ out, const float* __restrict__ a, int n) {
  int i = blockIdx.x * 256 + threadIdx.x;
  if (i >= n) return;
  float v = out[i];
  out[i] = (v >= 0.0f) ? v : a[0] * v;
}

// ---------------- host orchestration ----------------

extern "C" void kernel_launch(void* const* d_in, const int* in_sizes, int n_in,
                              void* d_out, int out_size, void* d_ws, size_t ws_size,
                              hipStream_t stream) {
  (void)in_sizes; (void)n_in; (void)out_size; (void)ws_size;
  const float* node_attr = (const float*)d_in[0];
  const float* edge_attr = (const float*)d_in[1];
  const int*   src       = (const int*)d_in[2];
  const int*   dst       = (const int*)d_in[3];
  const float* Wp        = (const float*)d_in[6];
  const float* bp        = (const float*)d_in[7];
  const float* Wg        = (const float*)d_in[8];
  const float* bg        = (const float*)d_in[9];
  const float* We        = (const float*)d_in[10];
  const float* be        = (const float*)d_in[11];
  const float* nncb      = (const float*)d_in[12];
  const float* gru_wih   = (const float*)d_in[13];
  const float* gru_whh   = (const float*)d_in[14];
  const float* gru_bih   = (const float*)d_in[15];
  const float* gru_bhh   = (const float*)d_in[16];
  const float* lstm_wih  = (const float*)d_in[17];
  const float* lstm_whh  = (const float*)d_in[18];
  const float* lstm_bih  = (const float*)d_in[19];
  const float* lstm_bhh  = (const float*)d_in[20];
  const float* Wsm       = (const float*)d_in[21];
  const float* bs        = (const float*)d_in[22];
  const float* prelu_a   = (const float*)d_in[23];
  float* out = (float*)d_out;

  constexpr int N = 10000, E = 40000, NB = 250;

  // workspace bump allocator
  char* base = (char*)d_ws;
  size_t off = 0;
  auto alloc = [&](size_t bytes) -> void* {
    void* p = base + off;
    off = (off + bytes + 255) & ~(size_t)255;
    return p;
  };
  _Float16* na16    = (_Float16*)alloc((size_t)N * 96 * 2);
  _Float16* wp16    = (_Float16*)alloc(64 * 96 * 2);
  _Float16* w2t16   = (_Float16*)alloc(64 * 896 * 2);
  _Float16* gwih16  = (_Float16*)alloc(192 * 64 * 2);
  _Float16* gwhh16  = (_Float16*)alloc(192 * 64 * 2);
  _Float16* lwih16  = (_Float16*)alloc(512 * 256 * 2);
  _Float16* lwhh16  = (_Float16*)alloc(512 * 128 * 2);
  _Float16* ws16    = (_Float16*)alloc(1024 * 256 * 2);
  float*    bsum    = (float*)alloc(512 * 4);
  float*    h0      = (float*)alloc((size_t)N * 64 * 4);
  float*    hcur    = (float*)alloc((size_t)N * 64 * 4);
  _Float16* hid16   = (_Float16*)alloc((size_t)N * 64 * 2);
  float*    deg     = (float*)alloc((size_t)N * 4);
  float*    egate   = (float*)alloc((size_t)E * 4);
  float*    agg     = (float*)alloc((size_t)N * 64 * 4);
  _Float16* x16     = (_Float16*)alloc((size_t)N * 64 * 2);
  float*    gi      = (float*)alloc((size_t)N * 192 * 4);
  float*    gh      = (float*)alloc((size_t)N * 192 * 4);
  float*    qstar   = (float*)alloc((size_t)NB * 256 * 4);
  _Float16* qstar16 = (_Float16*)alloc((size_t)NB * 256 * 2);
  float*    hl      = (float*)alloc((size_t)NB * 128 * 4);
  _Float16* hl16    = (_Float16*)alloc((size_t)NB * 128 * 2);
  float*    cl      = (float*)alloc((size_t)NB * 128 * 4);
  float*    g4      = (float*)alloc((size_t)NB * 512 * 4);

  auto grid1 = [](int n) { return dim3((unsigned)((n + 255) / 256)); };
  auto gemm = [&](const _Float16* A, const _Float16* W, const float* bias, float* C,
                  int M, int Nn, int Kp, int flags) {
    dim3 g((unsigned)(Nn / 64), (unsigned)((M + 15) / 16));
    k_wmma_gemm<<<g, 32, 0, stream>>>(A, W, bias, C, M, Nn, Kp, flags);
  };

  // ---- weight / input conversions ----
  k_cvt_pad_f16<<<grid1(N * 96), 256, 0, stream>>>(node_attr, na16, N, 74, 96);
  k_cvt_pad_f16<<<grid1(64 * 96), 256, 0, stream>>>(Wp, wp16, 64, 74, 96);
  k_build_w2t<<<grid1(64 * 896), 256, 0, stream>>>(We, be, w2t16);
  k_cvt_pad_f16<<<grid1(192 * 64), 256, 0, stream>>>(gru_wih, gwih16, 192, 64, 64);
  k_cvt_pad_f16<<<grid1(192 * 64), 256, 0, stream>>>(gru_whh, gwhh16, 192, 64, 64);
  k_cvt_pad_f16<<<grid1(512 * 256), 256, 0, stream>>>(lstm_wih, lwih16, 512, 256, 256);
  k_cvt_pad_f16<<<grid1(512 * 128), 256, 0, stream>>>(lstm_whh, lwhh16, 512, 128, 128);
  k_cvt_pad_f16<<<grid1(1024 * 256), 256, 0, stream>>>(Wsm, ws16, 1024, 256, 256);
  k_add_vec<<<grid1(512), 256, 0, stream>>>(lstm_bih, lstm_bhh, bsum, 512);

  // ---- node projection: h0 = relu(node_attr @ Wp^T + bp) ----
  gemm(na16, wp16, bp, h0, N, 64, 96, /*relu*/1);

  // ---- degrees + edge gate ----
  k_zero_f32<<<grid1(N), 256, 0, stream>>>(deg, N);
  k_deg<<<grid1(E), 256, 0, stream>>>(dst, deg, E);
  k_gate<<<grid1(E), 256, 0, stream>>>(h0, src, dst, Wg, bg, deg, egate, E);

  // ---- message passing (3 steps) ----
  k_copy_f32<<<grid1(N * 64), 256, 0, stream>>>(hcur, h0, N * 64);
  k_cvt_pad_f16<<<grid1(N * 64), 256, 0, stream>>>(hcur, hid16, N, 64, 64);
  for (int step = 0; step < 3; ++step) {
    k_zero_f32<<<grid1(N * 64), 256, 0, stream>>>(agg, N * 64);
    k_edge_msg<<<E / 32, 256, 0, stream>>>(hcur, edge_attr, egate, src, dst, w2t16, agg, E);
    k_relu_x16<<<grid1(N * 64), 256, 0, stream>>>(agg, nncb, x16, N * 64);
    gemm(x16, gwih16, gru_bih, gi, N, 192, 64, 0);
    gemm(hid16, gwhh16, gru_bhh, gh, N, 192, 64, 0);
    k_gru<<<grid1(N * 64), 256, 0, stream>>>(gi, gh, hcur, hid16, N * 64);
  }

  // ---- Set2Set readout ----
  k_zero_f32<<<grid1(NB * 256), 256, 0, stream>>>(qstar, NB * 256);
  k_zero_f16<<<grid1(NB * 256), 256, 0, stream>>>(qstar16, NB * 256);
  k_zero_f32<<<grid1(NB * 128), 256, 0, stream>>>(hl, NB * 128);
  k_zero_f16<<<grid1(NB * 128), 256, 0, stream>>>(hl16, NB * 128);
  k_zero_f32<<<grid1(NB * 128), 256, 0, stream>>>(cl, NB * 128);
  for (int step = 0; step < 3; ++step) {
    gemm(qstar16, lwih16, bsum, g4, NB, 512, 256, 0);
    gemm(hl16, lwhh16, nullptr, g4, NB, 512, 128, /*acc*/2);
    k_lstm<<<grid1(NB * 128), 256, 0, stream>>>(g4, cl, hl, hl16, NB * 128);
    k_attn<<<NB, 64, 0, stream>>>(h0, hcur, hl, qstar, qstar16);
  }

  // ---- sparsify: out = prelu(q_star @ Ws^T + bs) ----
  gemm(qstar16, ws16, bs, out, NB, 1024, 256, 0);
  k_prelu<<<grid1(NB * 1024), 256, 0, stream>>>(out, prelu_a, NB * 1024);
}
